// SpatialDGNN_7919919694132
// MI455X (gfx1250) — matrix operand order repaired and checked by reading, not compiled, not run on hardware
//
#include <hip/hip_runtime.h>

// GCN 2-layer pipeline for MI455X (gfx1250, wave32).
// GEMMs via v_wmma_f32_16x16x32_f16 with pre-packed zero-padded f16 operands
// (branch-free inner loop, all-b128 fragment loads).
// Message passing in fp32 with global_atomic_add_f32 (L2-resident: xw/h fit in 192MB L2).

typedef __attribute__((ext_vector_type(16))) _Float16 v16h;
typedef __attribute__((ext_vector_type(8)))  float    v8f;

union Frag {
  v16h  h;
  uint4 u[2];
};

#define IN_DIM  262
#define HIDDEN  512

// ---------------- utility ----------------
__global__ void zero_f32(float* __restrict__ p, long long n) {
  long long i = (long long)blockIdx.x * blockDim.x + threadIdx.x;
  long long stride = (long long)gridDim.x * blockDim.x;
  for (; i < n; i += stride) p[i] = 0.0f;
}

// deg[col] += w   (self-loop +1 folded into dinv_fin)
__global__ void deg_accum(float* __restrict__ deg, const long long* __restrict__ ei,
                          const float* __restrict__ ew, int E) {
  int e = blockIdx.x * blockDim.x + threadIdx.x;
  if (e >= E) return;
  int col = (int)ei[E + e];
  atomicAdd(&deg[col], ew[e]);
}

__global__ void dinv_fin(float* __restrict__ dinv, int N) {
  int i = blockIdx.x * blockDim.x + threadIdx.x;
  if (i >= N) return;
  dinv[i] = rsqrtf(dinv[i] + 1.0f);   // deg + self-loop weight(1) > 0 always
}

__global__ void norm_k(float* __restrict__ nrm, const float* __restrict__ dinv,
                       const long long* __restrict__ ei, const float* __restrict__ ew, int E) {
  int e = blockIdx.x * blockDim.x + threadIdx.x;
  if (e >= E) return;
  int row = (int)ei[e];
  int col = (int)ei[E + e];
  nrm[e] = dinv[row] * ew[e] * dinv[col];
}

// ---------------- operand staging (fp32 -> padded f16) ----------------
// A[M][K] fp32  ->  Ah[M][Kp] f16, zero fill k >= K   (Kp % 32 == 0)
__global__ void pack_a(_Float16* __restrict__ dst, const float* __restrict__ src,
                       int M, int K, int Kp) {
  int idx = blockIdx.x * blockDim.x + threadIdx.x;
  if (idx >= M * Kp) return;
  int r = idx / Kp;
  int k = idx - r * Kp;
  dst[idx] = (k < K) ? (_Float16)src[r * K + k] : (_Float16)0.0f;
}

// B[K][N] fp32  ->  Bt[Npad][Kp] f16 (transposed, K contiguous), zero fill
__global__ void pack_bt(_Float16* __restrict__ dst, const float* __restrict__ src,
                        int K, int N, int Kp, int Npad) {
  int idx = blockIdx.x * blockDim.x + threadIdx.x;
  if (idx >= Npad * Kp) return;
  int n = idx / Kp;
  int k = idx - n * Kp;
  dst[idx] = (k < K && n < N) ? (_Float16)src[k * N + n] : (_Float16)0.0f;
}

// ---------------- WMMA GEMM: C[M][N] = Ah[M][Kp] @ Bt[Npad][Kp]^T ----------------
// One wave computes a 16x64 strip of C (4 accumulators). Branch-free K loop:
// per chunk: 2x b128 A loads, 8x b128 B loads, 4x v_wmma_f32_16x16x32_f16.
// Requires: M % 16 == 0, Kp % 32 == 0, Npad % 64 == 0.
__global__ void gemm_wmma(const _Float16* __restrict__ Ah, const _Float16* __restrict__ Bt,
                          float* __restrict__ C, int M, int Kp, int N, int ldc, int nW4) {
  int lane = threadIdx.x & 31;
  int wave = (blockIdx.x * blockDim.x + threadIdx.x) >> 5;
  int Mt = M >> 4;
  if (wave >= Mt * nW4) return;          // wave-uniform exit
  int tm = wave / nW4;
  int t4 = wave - tm * nW4;
  int half = lane >> 4;                  // 0: lanes 0-15, 1: lanes 16-31
  int r    = lane & 15;

  const _Float16* ap = Ah + (tm * 16 + r) * Kp;
  const _Float16* bp0 = Bt + (t4 * 64 + 0  + r) * Kp;
  const _Float16* bp1 = Bt + (t4 * 64 + 16 + r) * Kp;
  const _Float16* bp2 = Bt + (t4 * 64 + 32 + r) * Kp;
  const _Float16* bp3 = Bt + (t4 * 64 + 48 + r) * Kp;

  v8f acc0 = {}, acc1 = {}, acc2 = {}, acc3 = {};
  int nkc = Kp >> 5;
  for (int kc = 0; kc < nkc; ++kc) {
    int k0 = kc << 5;
    Frag a;
    // A k order used here (same as B): lanes0-15 K={0..15}; lanes16-31 K={16..31}
    a.u[0] = *(const uint4*)(ap + k0 + (half << 4));
    a.u[1] = *(const uint4*)(ap + k0 + (half << 4) + 8);
    // B k order used here (same as A): lane n -> col n, K = j + 16*half (j = 0..15 contiguous)
    Frag b0, b1, b2, b3;
    int kb = k0 + (half << 4);
    b0.u[0] = *(const uint4*)(bp0 + kb);
    b0.u[1] = *(const uint4*)(bp0 + kb + 8);
    b1.u[0] = *(const uint4*)(bp1 + kb);
    b1.u[1] = *(const uint4*)(bp1 + kb + 8);
    b2.u[0] = *(const uint4*)(bp2 + kb);
    b2.u[1] = *(const uint4*)(bp2 + kb + 8);
    b3.u[0] = *(const uint4*)(bp3 + kb);
    b3.u[1] = *(const uint4*)(bp3 + kb + 8);
    acc0 = __builtin_amdgcn_wmma_f32_16x16x32_f16(false, a.h, false, b0.h, (short)0, acc0, false, false);
    acc1 = __builtin_amdgcn_wmma_f32_16x16x32_f16(false, a.h, false, b1.h, (short)0, acc1, false, false);
    acc2 = __builtin_amdgcn_wmma_f32_16x16x32_f16(false, a.h, false, b2.h, (short)0, acc2, false, false);
    acc3 = __builtin_amdgcn_wmma_f32_16x16x32_f16(false, a.h, false, b3.h, (short)0, acc3, false, false);
  }
  // C layout: VGPR v -> row v + 8*half, col = r (within tile)
#pragma unroll
  for (int v = 0; v < 8; ++v) {
    int ro = (tm << 4) + v + (half << 3);   // always < M (M % 16 == 0)
    float* crow = C + ro * ldc + t4 * 64 + r;
    int cb = t4 * 64 + r;
    if (cb      < N) crow[0]  = acc0[v];
    if (cb + 16 < N) crow[16] = acc1[v];
    if (cb + 32 < N) crow[32] = acc2[v];
    if (cb + 48 < N) crow[48] = acc3[v];
  }
}

// ---------------- scatter (gather src[row], scale, atomic-add into dst[col]) ----------------
// Layer1: F=512 -> 128 float4 chunks per edge
__global__ void scatter512(float* __restrict__ dst, const float* __restrict__ src,
                           const long long* __restrict__ ei, const float* __restrict__ nrm, int E) {
  int idx = blockIdx.x * blockDim.x + threadIdx.x;
  if (idx >= E * 128) return;
  int e = idx >> 7;
  int f = (idx & 127) << 2;
  int row = (int)ei[e];
  int col = (int)ei[E + e];
  float w = nrm[e];
  const float4 v = *(const float4*)(src + row * HIDDEN + f);
  float* d = dst + col * HIDDEN + f;
  atomicAdd(d + 0, w * v.x);
  atomicAdd(d + 1, w * v.y);
  atomicAdd(d + 2, w * v.z);
  atomicAdd(d + 3, w * v.w);
}

// Layer2: F=262 -> 131 float2 chunks per edge
__global__ void scatter262(float* __restrict__ dst, const float* __restrict__ src,
                           const long long* __restrict__ ei, const float* __restrict__ nrm, int E) {
  int idx = blockIdx.x * blockDim.x + threadIdx.x;
  const int C2 = 131;
  if (idx >= E * C2) return;
  int e = idx / C2;
  int f = (idx - e * C2) << 1;
  int row = (int)ei[e];
  int col = (int)ei[E + e];
  float w = nrm[e];
  const float2 v = *(const float2*)(src + row * IN_DIM + f);
  float* d = dst + col * IN_DIM + f;
  atomicAdd(d + 0, w * v.x);
  atomicAdd(d + 1, w * v.y);
}

// h = relu(agg + dinv[i]^2 * xw[i] + b1)   (self-loop folded in; in-place)
__global__ void bias_relu(float* __restrict__ h, const float* __restrict__ xw,
                          const float* __restrict__ dinv, const float* __restrict__ b, int N) {
  int idx = blockIdx.x * blockDim.x + threadIdx.x;
  if (idx >= N * HIDDEN) return;
  int i = idx >> 9;
  int f = idx & 511;
  float d2 = dinv[i] * dinv[i];
  float v = h[idx] + d2 * xw[idx] + b[f];
  h[idx] = v > 0.0f ? v : 0.0f;
}

// out += dinv[i]^2 * xw[i] + b2   (in-place on d_out holding edge aggregation)
__global__ void final_bias(float* __restrict__ out, const float* __restrict__ xw,
                           const float* __restrict__ dinv, const float* __restrict__ b, int N) {
  int idx = blockIdx.x * blockDim.x + threadIdx.x;
  if (idx >= N * IN_DIM) return;
  int i = idx / IN_DIM;
  int f = idx - i * IN_DIM;
  float d2 = dinv[i] * dinv[i];
  out[idx] = out[idx] + d2 * xw[idx] + b[f];
}

// ---------------- host orchestration ----------------
extern "C" void kernel_launch(void* const* d_in, const int* in_sizes, int n_in,
                              void* d_out, int out_size, void* d_ws, size_t ws_size,
                              hipStream_t stream) {
  const float*      x  = (const float*)d_in[0];
  const long long*  ei = (const long long*)d_in[1];   // int64 edge_index [2,E]
  const float*      ew = (const float*)d_in[2];
  const float*      W1 = (const float*)d_in[3];
  const float*      b1 = (const float*)d_in[4];
  const float*      W2 = (const float*)d_in[5];
  const float*      b2 = (const float*)d_in[6];
  float*            out = (float*)d_out;

  const int N = in_sizes[0] / IN_DIM;   // 50000 (multiple of 16)
  const int E = in_sizes[1] / 2;        // 400000

  const int KP1 = 288;   // 262 padded to 32
  const int KP2 = 512;   // already multiple of 32
  const int NP1 = 512;   // multiple of 64
  const int NP2 = 320;   // 262 padded to 64

  // workspace layout (16B-aligned chunks)
  size_t off = 0;
  auto alloc = [&](size_t bytes) {
    void* p = (char*)d_ws + off;
    off = (off + bytes + 15) & ~(size_t)15;
    return p;
  };
  float*    dinv = (float*)alloc((size_t)N * 4);
  float*    nrm  = (float*)alloc((size_t)E * 4);
  float*    bufA = (float*)alloc((size_t)N * HIDDEN * 4);           // xw1, then xw2 (fp32)
  float*    bufB = (float*)alloc((size_t)N * HIDDEN * 4);           // agg1 -> h (fp32)
  _Float16* hA   = (_Float16*)alloc((size_t)N * KP2 * 2);           // packed A (max of KP1/KP2)
  _Float16* hB   = (_Float16*)alloc((size_t)NP2 * KP2 * 2);         // packed Bt (max size)

  const int B = 256;  // 8 wave32s per workgroup
  auto blocks = [](long long work, int b) { return (int)((work + b - 1) / b); };

  // --- normalization ---
  zero_f32<<<blocks(N, B), B, 0, stream>>>(dinv, N);
  deg_accum<<<blocks(E, B), B, 0, stream>>>(dinv, ei, ew, E);
  dinv_fin<<<blocks(N, B), B, 0, stream>>>(dinv, N);
  norm_k<<<blocks(E, B), B, 0, stream>>>(nrm, dinv, ei, ew, E);

  // --- layer 1: xw1 = x @ W1 (WMMA), aggregate, bias+relu ---
  pack_a <<<blocks((long long)N * KP1, B), B, 0, stream>>>(hA, x, N, IN_DIM, KP1);
  pack_bt<<<blocks((long long)NP1 * KP1, B), B, 0, stream>>>(hB, W1, IN_DIM, HIDDEN, KP1, NP1);
  {
    long long waves = (long long)(N / 16) * (NP1 / 64);
    gemm_wmma<<<blocks(waves * 32, B), B, 0, stream>>>(hA, hB, bufA, N, KP1, HIDDEN, HIDDEN, NP1 / 64);
  }
  zero_f32<<<blocks((long long)N * HIDDEN, B), B, 0, stream>>>(bufB, (long long)N * HIDDEN);
  scatter512<<<blocks((long long)E * 128, B), B, 0, stream>>>(bufB, bufA, ei, nrm, E);
  bias_relu<<<blocks((long long)N * HIDDEN, B), B, 0, stream>>>(bufB, bufA, dinv, b1, N);

  // --- layer 2: xw2 = h @ W2 (WMMA), aggregate into d_out, bias ---
  pack_a <<<blocks((long long)N * KP2, B), B, 0, stream>>>(hA, bufB, N, HIDDEN, KP2);
  pack_bt<<<blocks((long long)NP2 * KP2, B), B, 0, stream>>>(hB, W2, HIDDEN, IN_DIM, KP2, NP2);
  {
    long long waves = (long long)(N / 16) * (NP2 / 64);
    gemm_wmma<<<blocks(waves * 32, B), B, 0, stream>>>(hA, hB, bufA, N, KP2, IN_DIM, IN_DIM, NP2 / 64);
  }
  zero_f32<<<blocks((long long)N * IN_DIM, B), B, 0, stream>>>(out, (long long)N * IN_DIM);
  scatter262<<<blocks((long long)E * 131, B), B, 0, stream>>>(out, bufA, ei, nrm, E);
  final_bias<<<blocks((long long)N * IN_DIM, B), B, 0, stream>>>(out, bufA, dinv, b2, N);
}
